// EGNN_37555194036730
// MI455X (gfx1250) — compile-verified, matrix operand order checked by reading, not run on hardware
//
#include <hip/hip_runtime.h>
#include <hip/hip_bf16.h>

#define DD 128
#define HH 8

typedef __bf16 bf16_t;
typedef __attribute__((ext_vector_type(16))) __bf16 v16bf;
typedef __attribute__((ext_vector_type(8)))  __bf16 v8bf;
typedef __attribute__((ext_vector_type(4)))  __bf16 v4bf;
typedef __attribute__((ext_vector_type(8)))  float  v8f;

union V16U { v16bf v; v8bf h[2]; };

__device__ __forceinline__ bf16_t f2bf(float x) { return (bf16_t)x; }

__device__ __forceinline__ v4bf pk4(float4 v) {
  v4bf o; o[0] = f2bf(v.x); o[1] = f2bf(v.y); o[2] = f2bf(v.z); o[3] = f2bf(v.w);
  return o;
}

// ---- weight pack: row-major f32 [K,N] -> per-lane B fragments ------------
// tile index = nt*numKT + kt ; within tile: 32 lanes x 16 bf16
// mode 1: E_GCL input permutation — packed row k' reads reference row:
//   k'<256 -> k' ; 256<=k'<384 -> k'+1 ; k'==384 -> 256 ; else zero-pad
__global__ void k_pack(const float* __restrict__ W, bf16_t* __restrict__ out,
                       int K, int Kpad, int N, int mode) {
  int numKT = Kpad >> 5;
  int total = numKT * (N >> 4) * 32;
  int id = blockIdx.x * blockDim.x + threadIdx.x;
  if (id >= total) return;
  int lane = id & 31;
  int tile = id >> 5;
  int kt = tile % numKT;
  int nt = tile / numKT;
  int n = (nt << 4) + (lane & 15);
  int half = lane >> 4;
  bf16_t* o = out + ((size_t)tile * 32 + lane) * 16;
  for (int j = 0; j < 16; ++j) {
    int k = (kt << 5) + ((j < 8) ? (half * 8 + j) : (16 + half * 8 + (j - 8)));
    int kr;
    if (mode == 1) {
      if (k < 256)      kr = k;
      else if (k < 384) kr = k + 1;
      else if (k == 384) kr = 256;
      else              kr = -1;
    } else {
      kr = (k < K) ? k : -1;
    }
    float v = (kr >= 0 && kr < K) ? W[(size_t)kr * N + n] : 0.0f;
    o[j] = f2bf(v);
  }
}

// ---- generic WMMA GEMM: C[M,N] = act(A[M,K] @ W + bias) ------------------
// One wave computes a 16x128 output tile (8 accumulators); A streamed once
// per 128-col group; next A K-slice prefetched (global_prefetch_b8).
template<int ACT, typename OUT_T>
__global__ void k_gemm(const bf16_t* __restrict__ A, int lda,
                       const bf16_t* __restrict__ Wp,
                       const float* __restrict__ bias,
                       OUT_T* __restrict__ C, int ldc,
                       int M, int K, int N) {
  const int lane = threadIdx.x & 31;
  const int wave = blockIdx.x * (blockDim.x >> 5) + (threadIdx.x >> 5);
  const int mTiles  = M >> 4;
  const int nGroups = (N + 127) >> 7;
  if (wave >= mTiles * nGroups) return;
  const int mTile = wave % mTiles;   // consecutive waves share ng -> B reuse in L0/L2
  const int ng    = wave / mTiles;
  const int numKT = K >> 5;
  const int numNT = N >> 4;
  const int nt0   = ng << 3;
  int ntCnt = numNT - nt0; if (ntCnt > 8) ntCnt = 8;
  const int mBase = mTile << 4;

  v8f acc[8];
  for (int t = 0; t < 8; ++t)
    for (int r = 0; r < 8; ++r) acc[t][r] = 0.0f;

  const int m    = mBase + (lane & 15);
  const int half = lane >> 4;
  const bf16_t* arow = A + (size_t)m * lda + half * 8;

  for (int kt = 0; kt < numKT; ++kt) {
    V16U u;
    u.h[0] = *(const v8bf*)(arow);
    u.h[1] = *(const v8bf*)(arow + 16);
    if (kt + 1 < numKT) __builtin_prefetch(arow + 32, 0, 1);
    arow += 32;
    const v16bf* bp = ((const v16bf*)Wp) + ((size_t)nt0 * numKT + kt) * 32 + lane;
    for (int t = 0; t < ntCnt; ++t) {
      v16bf b = bp[(size_t)t * numKT * 32];
      acc[t] = __builtin_amdgcn_wmma_f32_16x16x32_bf16(false, u.v, false, b,
                                                       (short)0, acc[t], false, false);
    }
  }
  const int col0  = lane & 15;
  const int halfm = lane >> 4;
  for (int t = 0; t < ntCnt; ++t) {
    int col = ((nt0 + t) << 4) + col0;
    float bv = bias ? bias[col] : 0.0f;
    for (int r = 0; r < 8; ++r) {
      float x = acc[t][r] + bv;
      if (ACT == 1) x = fmaxf(x, 0.0f);
      if (ACT == 2) x = x / (1.0f + __expf(-x));
      int row = mBase + r + (halfm << 3);
      C[(size_t)row * ldc + col] = (OUT_T)x;
    }
  }
}

// ---- fused FAN attention: per (edge,head) 32->32 relu ->16, softmax, *v, scatter
__global__ void k_fan_att(const bf16_t* __restrict__ qB, const bf16_t* __restrict__ kB,
                          const float* __restrict__ vF,
                          const bf16_t* __restrict__ W1p, const float* __restrict__ b1,
                          const bf16_t* __restrict__ W2p, const float* __restrict__ b2,
                          const int* __restrict__ src, float* __restrict__ agg,
                          int E, float invTemp) {
  __shared__ __align__(16) char smemRaw[8 * 1024];
  const int lane = threadIdx.x & 31;
  const int wib  = threadIdx.x >> 5;
  const int wave = blockIdx.x * 8 + wib;
  const int totalRows = E * HH;
  const int rowBase = wave << 4;
  if (rowBase >= totalRows) return;
  char* my = smemRaw + wib * 1024;
  bf16_t* hid = (bf16_t*)my;   // [16][32] bf16
  float*  sc  = (float*)my;    // [16][16] f32 (reused after hid consumed)

  const int lrow = lane & 15;
  const int half = lane >> 4;
  const int grow = rowBase + lrow;
  const int e  = grow >> 3;
  const int hd = grow & 7;

  // A fragment: row (e,hd): cols 0..15 = q[e, c*8+hd], cols 16..31 = k[e, c*8+hd]
  V16U ua;
#pragma unroll
  for (int j = 0; j < 16; ++j) {
    int Kk = (j < 8) ? (half * 8 + j) : (16 + half * 8 + (j - 8));
    bf16_t val = (Kk < 16) ? qB[(size_t)e * DD + Kk * HH + hd]
                           : kB[(size_t)e * DD + (Kk - 16) * HH + hd];
    ua.v[j] = val;
  }
  // hidden = relu(A @ W1 + b1)  (16x32)
  v8f h0, h1;
  for (int r = 0; r < 8; ++r) { h0[r] = 0.0f; h1[r] = 0.0f; }
  v16bf bw0 = ((const v16bf*)W1p)[0 * 32 + lane];
  v16bf bw1 = ((const v16bf*)W1p)[1 * 32 + lane];
  h0 = __builtin_amdgcn_wmma_f32_16x16x32_bf16(false, ua.v, false, bw0, (short)0, h0, false, false);
  h1 = __builtin_amdgcn_wmma_f32_16x16x32_bf16(false, ua.v, false, bw1, (short)0, h1, false, false);
  {
    float bv0 = b1[lrow], bv1 = b1[16 + lrow];
    for (int r = 0; r < 8; ++r) {
      int row = r + (half << 3);
      hid[row * 32 + lrow]      = f2bf(fmaxf(h0[r] + bv0, 0.0f));
      hid[row * 32 + 16 + lrow] = f2bf(fmaxf(h1[r] + bv1, 0.0f));
    }
  }
  // re-fragment hidden (C-layout -> A-layout) through per-wave LDS
  V16U ua2;
#pragma unroll
  for (int j = 0; j < 16; ++j) {
    int Kk = (j < 8) ? (half * 8 + j) : (16 + half * 8 + (j - 8));
    ua2.v[j] = hid[lrow * 32 + Kk];
  }
  v8f s;
  for (int r = 0; r < 8; ++r) s[r] = 0.0f;
  v16bf bw2 = ((const v16bf*)W2p)[lane];
  s = __builtin_amdgcn_wmma_f32_16x16x32_bf16(false, ua2.v, false, bw2, (short)0, s, false, false);
  {
    float bv = b2[lrow];
    for (int r = 0; r < 8; ++r) {
      int row = r + (half << 3);
      sc[row * 16 + lrow] = (s[r] + bv) * invTemp;
    }
  }
  // softmax over 16 channels, multiply by v, atomic scatter to agg[src]
  if (lane < 16) {
    int g2 = rowBase + lane;
    int e2 = g2 >> 3, h2 = g2 & 7;
    float buf[16], mx = -1e30f;
    for (int c = 0; c < 16; ++c) { buf[c] = sc[lane * 16 + c]; mx = fmaxf(mx, buf[c]); }
    float sum = 0.0f;
    for (int c = 0; c < 16; ++c) { buf[c] = __expf(buf[c] - mx); sum += buf[c]; }
    float inv = 1.0f / sum;
    int s0 = src[e2];
    float* dstp = agg + (size_t)s0 * DD;
    const float* vrow = vF + (size_t)e2 * DD;
    for (int c = 0; c < 16; ++c)
      atomicAdd(dstp + c * HH + h2, buf[c] * inv * vrow[c * HH + h2]);
  }
}

// ---- vectorized gather / concat / scatter helpers ------------------------
// msg_fan input: [h[src] | edge_attr | h[dst]]  (E x 384 bf16)
__global__ void k_gather3(const float* __restrict__ h, const float* __restrict__ ea,
                          const int* __restrict__ src, const int* __restrict__ dst,
                          bf16_t* __restrict__ T, int E) {
  long long i = (long long)blockIdx.x * blockDim.x + threadIdx.x;
  if (i >= (long long)E * 96) return;
  int e = (int)(i / 96);
  int j = (int)(i % 96) * 4;
  const float4* p;
  if (j < 128)      p = (const float4*)(h  + (size_t)src[e] * DD + j);
  else if (j < 256) p = (const float4*)(ea + (size_t)e * DD + (j - 128));
  else              p = (const float4*)(h  + (size_t)dst[e] * DD + (j - 256));
  *(v4bf*)(T + (size_t)e * 384 + j) = pk4(*p);
}

// e_gcl input (permuted layout, matches k_pack mode 1):
// [h[src](0-127) | h[dst](128-255) | edge_attr(256-383) | radial(384) | pad]
__global__ void k_egcl_in(const float* __restrict__ h, const float* __restrict__ ea,
                          const float* __restrict__ coord,
                          const int* __restrict__ src, const int* __restrict__ dst,
                          bf16_t* __restrict__ T, float* __restrict__ cdiff, int E) {
  long long i = (long long)blockIdx.x * blockDim.x + threadIdx.x;
  if (i >= (long long)E * 104) return;
  int e = (int)(i / 104);
  int j = (int)(i % 104) * 4;
  v4bf o;
  if (j < 384) {
    const float4* p;
    if (j < 128)      p = (const float4*)(h  + (size_t)src[e] * DD + j);
    else if (j < 256) p = (const float4*)(h  + (size_t)dst[e] * DD + (j - 128));
    else              p = (const float4*)(ea + (size_t)e * DD + (j - 256));
    o = pk4(*p);
  } else if (j == 384) {
    int s0 = src[e], d0 = dst[e];
    float dx = coord[(size_t)s0 * 3 + 0] - coord[(size_t)d0 * 3 + 0];
    float dy = coord[(size_t)s0 * 3 + 1] - coord[(size_t)d0 * 3 + 1];
    float dz = coord[(size_t)s0 * 3 + 2] - coord[(size_t)d0 * 3 + 2];
    cdiff[(size_t)e * 3 + 0] = dx;
    cdiff[(size_t)e * 3 + 1] = dy;
    cdiff[(size_t)e * 3 + 2] = dz;
    o[0] = f2bf(dx * dx + dy * dy + dz * dz);
    o[1] = o[2] = o[3] = f2bf(0.0f);
  } else {
    o[0] = o[1] = o[2] = o[3] = f2bf(0.0f);
  }
  *(v4bf*)(T + (size_t)e * 416 + j) = o;
}

__global__ void k_concat2(const float* __restrict__ X, const float* __restrict__ Y,
                          bf16_t* __restrict__ T, int Nn) {
  long long i = (long long)blockIdx.x * blockDim.x + threadIdx.x;
  if (i >= (long long)Nn * 64) return;
  int n = (int)(i >> 6);
  int j = (int)(i & 63) * 4;
  const float4* p = (j < 128) ? (const float4*)(X + (size_t)n * DD + j)
                              : (const float4*)(Y + (size_t)n * DD + (j - 128));
  *(v4bf*)(T + (size_t)n * 256 + j) = pk4(*p);
}

__global__ void k_gate(const float* __restrict__ ef, const float* __restrict__ w,
                       const float* __restrict__ b, float* __restrict__ outF,
                       bf16_t* __restrict__ outB, int E) {
  int e = blockIdx.x * blockDim.x + threadIdx.x;
  if (e >= E) return;
  const float4* row = (const float4*)(ef + (size_t)e * DD);
  const float4* wv  = (const float4*)w;
  float s = b[0];
  for (int j = 0; j < DD / 4; ++j) {
    float4 r = row[j], ww = wv[j];
    s += r.x * ww.x + r.y * ww.y + r.z * ww.z + r.w * ww.w;
  }
  s = 1.0f / (1.0f + __expf(-s));
  float4* of = (float4*)(outF + (size_t)e * DD);
  v4bf*   ob = (v4bf*)(outB + (size_t)e * DD);
  for (int j = 0; j < DD / 4; ++j) {
    float4 r = row[j];
    r.x *= s; r.y *= s; r.z *= s; r.w *= s;
    of[j] = r;
    ob[j] = pk4(r);
  }
}

__global__ void k_cmsg(const float* __restrict__ c1o, const float* __restrict__ c2w,
                       const float* __restrict__ cdiff, const int* __restrict__ src,
                       float* __restrict__ cacc, int E) {
  int e = blockIdx.x * blockDim.x + threadIdx.x;
  if (e >= E) return;
  const float4* row = (const float4*)(c1o + (size_t)e * DD);
  const float4* wv  = (const float4*)c2w;
  float cm = 0.0f;
  for (int j = 0; j < DD / 4; ++j) {
    float4 r = row[j], ww = wv[j];
    cm += r.x * ww.x + r.y * ww.y + r.z * ww.z + r.w * ww.w;
  }
  int s0 = src[e];
  for (int t = 0; t < 3; ++t)
    atomicAdd(&cacc[(size_t)s0 * 3 + t], cdiff[(size_t)e * 3 + t] * cm);
}

__global__ void k_cupd(float* __restrict__ coord, const float* __restrict__ cacc,
                       const float* __restrict__ cnt, int Nn) {
  long long i = (long long)blockIdx.x * blockDim.x + threadIdx.x;
  if (i >= (long long)Nn * 3) return;
  int n = (int)(i / 3);
  coord[i] += cacc[i] / fmaxf(cnt[n], 1.0f);
}

__global__ void k_scat(const float* __restrict__ ef, const int* __restrict__ src,
                       float* __restrict__ agg, int E) {
  long long i = (long long)blockIdx.x * blockDim.x + threadIdx.x;
  if (i >= (long long)E * 32) return;
  int e = (int)(i >> 5);
  int j = (int)(i & 31) * 4;
  float4 v = *(const float4*)(ef + (size_t)e * DD + j);
  float* dstp = agg + (size_t)src[e] * DD + j;
  atomicAdd(dstp + 0, v.x);
  atomicAdd(dstp + 1, v.y);
  atomicAdd(dstp + 2, v.z);
  atomicAdd(dstp + 3, v.w);
}

__global__ void k_count(const int* __restrict__ src, float* __restrict__ cnt, int E) {
  int e = blockIdx.x * blockDim.x + threadIdx.x;
  if (e < E) atomicAdd(&cnt[src[e]], 1.0f);
}

__global__ void k_cast_bf16(const float* __restrict__ in, bf16_t* __restrict__ out, long long n4) {
  long long i = (long long)blockIdx.x * blockDim.x + threadIdx.x;
  if (i < n4) ((v4bf*)out)[i] = pk4(((const float4*)in)[i]);
}

__global__ void k_copyf(const float* __restrict__ in, float* __restrict__ out, long long n) {
  long long i = (long long)blockIdx.x * blockDim.x + threadIdx.x;
  if (i < n) out[i] = in[i];
}

__global__ void k_copy4(const float* __restrict__ in, float* __restrict__ out, long long n4) {
  long long i = (long long)blockIdx.x * blockDim.x + threadIdx.x;
  if (i < n4) ((float4*)out)[i] = ((const float4*)in)[i];
}

__global__ void k_addf4(float* __restrict__ a, const float* __restrict__ b, long long n4) {
  long long i = (long long)blockIdx.x * blockDim.x + threadIdx.x;
  if (i < n4) {
    float4 x = ((float4*)a)[i];
    float4 y = ((const float4*)b)[i];
    x.x += y.x; x.y += y.y; x.z += y.z; x.w += y.w;
    ((float4*)a)[i] = x;
  }
}

// ==========================================================================
extern "C" void kernel_launch(void* const* d_in, const int* in_sizes, int n_in,
                              void* d_out, int out_size, void* d_ws, size_t ws_size,
                              hipStream_t stream) {
  (void)n_in; (void)out_size; (void)ws_size;
  const int Nn = in_sizes[0] / DD;   // 50000
  const int E  = in_sizes[3] / 2;    // 400000
  const float* h_in     = (const float*)d_in[0];
  const float* coord_in = (const float*)d_in[1];
  const float* ea_in    = (const float*)d_in[2];
  const int*   ei       = (const int*)d_in[3];
  const int* src = ei;
  const int* dst = ei + E;
  auto P = [&](int i) -> const float* { return (const float*)d_in[i]; };

  // ---- workspace allocator ----
  char* wp = (char*)d_ws;
  auto alloc = [&](size_t bytes) -> char* {
    char* r = wp; wp += (bytes + 255) & ~(size_t)255; return r;
  };
  auto packW = [&](const float* W, int K, int Kpad, int N, int mode) -> bf16_t* {
    bf16_t* o = (bf16_t*)alloc((size_t)Kpad * N * 2);
    int total = (Kpad / 32) * (N / 16) * 32;
    k_pack<<<(total + 255) / 256, 256, 0, stream>>>(W, o, K, Kpad, N, mode);
    return o;
  };

  // ---- pack weights (leaf order: h,coord,edge_attr,edge_index, emb_in(W,b),
  //      emb_out(W,b), then per layer q,k,v,att1,att2,edge1,edge2,upd1,upd2,
  //      gcl_e1,gcl_e2,gcl_att,gcl_c1,gcl_c2(W only),gcl_n1,gcl_n2) ----
  bf16_t* pEmbIn  = packW(P(4), 128, 128, 128, 0);
  bf16_t* pEmbOut = packW(P(6), 128, 128, 128, 0);
  struct LayerP { bf16_t *q, *k, *v, *a1, *a2, *e1, *e2, *u1, *u2, *ge1, *ge2, *gc1, *gn1, *gn2; };
  LayerP LP[2];
  for (int l = 0; l < 2; ++l) {
    int b = 8 + l * 31;
    LP[l].q  = packW(P(b + 0), 128, 128, 128, 0);
    LP[l].k  = packW(P(b + 2), 128, 128, 128, 0);
    LP[l].v  = packW(P(b + 4), 128, 128, 128, 0);
    LP[l].a1 = packW(P(b + 6), 32, 32, 32, 0);
    LP[l].a2 = packW(P(b + 8), 32, 32, 16, 0);
    LP[l].e1 = packW(P(b + 10), 384, 384, 256, 0);
    LP[l].e2 = packW(P(b + 12), 256, 256, 128, 0);
    LP[l].u1 = packW(P(b + 14), 256, 256, 256, 0);
    LP[l].u2 = packW(P(b + 16), 256, 256, 128, 0);
    LP[l].ge1 = packW(P(b + 18), 385, 416, 128, 1);  // permuted rows (radial last)
    LP[l].ge2 = packW(P(b + 20), 128, 128, 128, 0);
    LP[l].gc1 = packW(P(b + 24), 128, 128, 128, 0);
    LP[l].gn1 = packW(P(b + 27), 256, 256, 128, 0);
    LP[l].gn2 = packW(P(b + 29), 128, 128, 128, 0);
  }

  // ---- activation buffers ----
  float*  hN       = (float*)alloc((size_t)Nn * DD * 4);
  bf16_t* hBF      = (bf16_t*)alloc((size_t)Nn * DD * 2);
  float*  edgeW    = (float*)alloc((size_t)E * DD * 4);
  float*  cnt      = (float*)alloc((size_t)Nn * 4);
  float*  aggN     = (float*)alloc((size_t)Nn * DD * 4);
  float*  coordW   = (float*)alloc((size_t)Nn * 3 * 4);
  float*  coordAcc = (float*)alloc((size_t)Nn * 3 * 4);
  float*  cdiff    = (float*)alloc((size_t)E * 3 * 4);
  bf16_t* Tbig     = (bf16_t*)alloc((size_t)E * 416 * 2);
  bf16_t* t1       = (bf16_t*)alloc((size_t)E * 256 * 2);
  bf16_t* qBF      = (bf16_t*)alloc((size_t)E * DD * 2);
  bf16_t* kBF      = (bf16_t*)alloc((size_t)E * DD * 2);
  float*  vF       = (float*)alloc((size_t)E * DD * 4);
  bf16_t* efBF     = (bf16_t*)alloc((size_t)E * DD * 2);
  bf16_t* hcat     = (bf16_t*)alloc((size_t)Nn * 256 * 2);
  bf16_t* nmid     = (bf16_t*)alloc((size_t)Nn * DD * 2);
  float*  hnT      = (float*)alloc((size_t)Nn * DD * 4);
  float*  efF      = (float*)qBF;  // alias: q/k dead during E_GCL; E*DD*2 is 256B-multiple
  float*  c1out    = vF;           // alias: v dead during E_GCL

  auto gemm = [&](int act, const bf16_t* A, int lda, const bf16_t* Wp, const float* bias,
                  void* C, bool bfOut, int ldc, int M, int K, int N) {
    int waves = (M / 16) * ((N + 127) / 128);
    int blocks = (waves + 7) / 8;
    if (!bfOut) {
      if (act == 0)      k_gemm<0, float><<<blocks, 256, 0, stream>>>(A, lda, Wp, bias, (float*)C, ldc, M, K, N);
      else if (act == 1) k_gemm<1, float><<<blocks, 256, 0, stream>>>(A, lda, Wp, bias, (float*)C, ldc, M, K, N);
      else               k_gemm<2, float><<<blocks, 256, 0, stream>>>(A, lda, Wp, bias, (float*)C, ldc, M, K, N);
    } else {
      if (act == 0)      k_gemm<0, bf16_t><<<blocks, 256, 0, stream>>>(A, lda, Wp, bias, (bf16_t*)C, ldc, M, K, N);
      else if (act == 1) k_gemm<1, bf16_t><<<blocks, 256, 0, stream>>>(A, lda, Wp, bias, (bf16_t*)C, ldc, M, K, N);
      else               k_gemm<2, bf16_t><<<blocks, 256, 0, stream>>>(A, lda, Wp, bias, (bf16_t*)C, ldc, M, K, N);
    }
  };
  auto GB = [](long long n) { return dim3((unsigned)((n + 255) / 256)); };

  // ---- prologue ----
  hipMemsetAsync(cnt, 0, (size_t)Nn * 4, stream);
  k_count<<<GB(E), 256, 0, stream>>>(src, cnt, E);
  k_copyf<<<GB((long long)Nn * 3), 256, 0, stream>>>(coord_in, coordW, (long long)Nn * 3);
  k_cast_bf16<<<GB((long long)Nn * 32), 256, 0, stream>>>(h_in, hBF, (long long)Nn * 32);
  gemm(0, hBF, 128, pEmbIn, P(5), hN, false, 128, Nn, 128, 128);   // h = emb_in(h)

  const float* curEdge = ea_in;
  for (int l = 0; l < 2; ++l) {
    int b = 8 + l * 31;
    int lastAct = (l == 0) ? 1 : 0;   // relu between layers only

    // -------- MSG_FAN --------
    k_gather3<<<GB((long long)E * 96), 256, 0, stream>>>(hN, curEdge, src, dst, Tbig, E);
    gemm(0, Tbig + 0,   384, LP[l].q, P(b + 1), qBF, true,  128, E, 128, 128);  // q(x_i)
    gemm(0, Tbig + 128, 384, LP[l].k, P(b + 3), kBF, true,  128, E, 128, 128);  // k(edge)
    gemm(0, Tbig + 256, 384, LP[l].v, P(b + 5), vF,  false, 128, E, 128, 128);  // v(x_j)
    gemm(1, Tbig, 384, LP[l].e1, P(b + 11), t1, true, 256, E, 384, 256);        // edge1+relu
    gemm(lastAct, t1, 256, LP[l].e2, P(b + 13), edgeW, false, 128, E, 256, 128);// t (new edge)
    hipMemsetAsync(aggN, 0, (size_t)Nn * DD * 4, stream);
    k_fan_att<<<(E / 2 + 7) / 8, 256, 0, stream>>>(qBF, kBF, vF, LP[l].a1, P(b + 7),
                                                   LP[l].a2, P(b + 9), src, aggN, E, 0.25f);
    k_concat2<<<GB((long long)Nn * 64), 256, 0, stream>>>(hN, aggN, hcat, Nn);
    gemm(1, hcat, 256, LP[l].u1, P(b + 15), t1, true, 256, Nn, 256, 256);       // upd1+relu
    gemm(lastAct, t1, 256, LP[l].u2, P(b + 17), hN, false, 128, Nn, 256, 128);  // h = upd2

    // -------- E_GCL --------
    k_egcl_in<<<GB((long long)E * 104), 256, 0, stream>>>(hN, edgeW, coordW, src, dst, Tbig, cdiff, E);
    gemm(2, Tbig, 416, LP[l].ge1, P(b + 19), t1, true, 128, E, 416, 128);       // silu
    gemm(2, t1, 128, LP[l].ge2, P(b + 21), efF, false, 128, E, 128, 128);       // silu
    k_gate<<<GB(E), 256, 0, stream>>>(efF, P(b + 22), P(b + 23), edgeW, efBF, E);
    gemm(2, efBF, 128, LP[l].gc1, P(b + 25), c1out, false, 128, E, 128, 128);   // silu
    hipMemsetAsync(coordAcc, 0, (size_t)Nn * 3 * 4, stream);
    k_cmsg<<<GB(E), 256, 0, stream>>>(c1out, P(b + 26), cdiff, src, coordAcc, E);
    k_cupd<<<GB((long long)Nn * 3), 256, 0, stream>>>(coordW, coordAcc, cnt, Nn);
    hipMemsetAsync(aggN, 0, (size_t)Nn * DD * 4, stream);
    k_scat<<<GB((long long)E * 32), 256, 0, stream>>>(edgeW, src, aggN, E);
    k_concat2<<<GB((long long)Nn * 64), 256, 0, stream>>>(hN, aggN, hcat, Nn);
    gemm(2, hcat, 256, LP[l].gn1, P(b + 28), nmid, true, 128, Nn, 256, 128);    // silu
    gemm(0, nmid, 128, LP[l].gn2, P(b + 30), hnT, false, 128, Nn, 128, 128);
    k_addf4<<<GB((long long)Nn * 32), 256, 0, stream>>>(hN, hnT, (long long)Nn * 32); // residual

    curEdge = edgeW;
  }

  // ---- epilogue: out = [emb_out(h), coord, edge_attr] ----
  float* out = (float*)d_out;
  k_cast_bf16<<<GB((long long)Nn * 32), 256, 0, stream>>>(hN, hBF, (long long)Nn * 32);
  gemm(0, hBF, 128, pEmbOut, P(7), out, false, 128, Nn, 128, 128);
  k_copyf<<<GB((long long)Nn * 3), 256, 0, stream>>>(coordW, out + (size_t)Nn * DD, (long long)Nn * 3);
  k_copy4<<<GB((long long)E * 32), 256, 0, stream>>>(edgeW, out + (size_t)Nn * DD + (size_t)Nn * 3,
                                                     (long long)E * 32);
}
